// MultiHeadAttention_56444460204440
// MI455X (gfx1250) — matrix operand order repaired and checked by reading, compiled, not run on hardware
//
#include <hip/hip_runtime.h>
#include <hip/hip_bf16.h>

// ---------------------------------------------------------------------------
// MI455X (gfx1250) multi-head causal attention, bf16 WMMA + fp32 accumulate.
//   q/k/v proj  : gemm_nt<0/1>  TDM (tensor_load_to_lds) staged, LDS-fed WMMA
//   attention   : flash_attn    TDM-staged K/V tiles, online softmax
//   out proj    : gemm_nt<2>    (+bias, fp32 output)
// ---------------------------------------------------------------------------

typedef __attribute__((ext_vector_type(16))) __bf16 v16bf;
typedef __attribute__((ext_vector_type(8)))  __bf16 v8bf;
typedef __attribute__((ext_vector_type(4)))  __bf16 v4bf;
typedef __attribute__((ext_vector_type(8)))  float  v8f;
typedef __attribute__((ext_vector_type(4)))  float  v4f;
typedef __attribute__((ext_vector_type(4)))  unsigned int u32x4;
typedef __attribute__((ext_vector_type(4)))  int      i32x4;
typedef __attribute__((ext_vector_type(8)))  int      i32x8;

union FragU { v16bf v; v8bf h[2]; };

// A-matrix 16x32 bf16 fragment (ISA 7.12.2): lane holds row M=lane&15.
// lane half 0: elems 0..7 <- K 0..7,  elems 8..15 <- K 16..23
// lane half 1: elems 0..7 <- K 8..15, elems 8..15 <- K 24..31
__device__ __forceinline__ v16bf load_a32(const __bf16* rowk, int lanehi) {
    FragU u;
    const int off = lanehi ? 8 : 0;
    u.h[0] = *(const v8bf*)(rowk + off);
    u.h[1] = *(const v8bf*)(rowk + off + 16);
    return u.v;
}

// B-matrix 32x16 bf16 fragment: lane holds column N=lane&15.
// lane half 0: elems 0..7 <- K 0..7,  elems 8..15 <- K 16..23
// lane half 1: elems 0..7 <- K 8..15, elems 8..15 <- K 24..31
__device__ __forceinline__ v16bf load_b32(const __bf16* colk, int lanehi) {
    FragU u;
    const int off = lanehi ? 8 : 0;
    u.h[0] = *(const v8bf*)(colk + off);
    u.h[1] = *(const v8bf*)(colk + off + 16);
    return u.v;
}

__device__ __forceinline__ v8f wmma_bf16(v16bf a, v16bf b, v8f c) {
    return __builtin_amdgcn_wmma_f32_16x16x32_bf16(
        false, a, false, b, (short)0, c, false, false);
}

// ---------------------------------------------------------------------------
// TDM: issue one 2D tile load (tile1 rows x tile0 elems, bf16) into LDS.
// pad_code: pad_interval code (4 -> every 128B, 5 -> every 256B); pad amount
// fixed at 16B, giving padded LDS row strides of 144B / 272B respectively.
// ---------------------------------------------------------------------------
#define TDM_AVAILABLE (__has_builtin(__builtin_amdgcn_tensor_load_to_lds))

#if TDM_AVAILABLE
__device__ __forceinline__ void tdm_load_2d(
    unsigned int lds_off, const __bf16* gaddr,
    unsigned int tile0, unsigned int tile1,
    unsigned int dim0, unsigned int dim1, unsigned int stride0,
    unsigned int pad_code) {
    const unsigned long long ga = (unsigned long long)(uintptr_t)gaddr;
    u32x4 g0;
    g0[0] = 1u;                                   // count=1, user descriptor
    g0[1] = lds_off;                              // LDS byte address
    g0[2] = (unsigned int)ga;                     // global_addr[31:0]
    g0[3] = (unsigned int)(ga >> 32) | (2u << 30);// global_addr[56:32] | type=2
    i32x8 g1;
    // data_size=2B (code1) | pad_enable | pad_interval | pad_amount=code3(16B)
    g1[0] = (int)((1u << 16) | (1u << 20) | (pad_code << 22) | (3u << 25));
    g1[1] = (int)((dim0 & 0xffffu) << 16);                    // tensor_dim0 lo
    g1[2] = (int)((dim0 >> 16) | ((dim1 & 0xffffu) << 16));   // dim0 hi|dim1 lo
    g1[3] = (int)((dim1 >> 16) | (tile0 << 16));              // dim1 hi|tile0
    g1[4] = (int)tile1;                                       // tile1 | tile2=0
    g1[5] = (int)stride0;                                     // dim0_stride lo
    g1[6] = 0;                                                // stride hi
    g1[7] = 0;
    const i32x4 z4 = {0, 0, 0, 0};
#if defined(__clang_major__) && (__clang_major__ >= 23)
    const i32x8 z8 = {0, 0, 0, 0, 0, 0, 0, 0};
    __builtin_amdgcn_tensor_load_to_lds(g0, g1, z4, z4, z8, 0);
#else
    __builtin_amdgcn_tensor_load_to_lds(g0, g1, z4, z4, 0);
#endif
}
#endif

// ---------------------------------------------------------------------------
// fp32 -> bf16 conversion (vectorized, grid-stride)
// ---------------------------------------------------------------------------
__global__ __launch_bounds__(256) void cvt_f32_bf16(
    const v4f* __restrict__ in, v4bf* __restrict__ out, int n4) {
    int i = blockIdx.x * blockDim.x + threadIdx.x;
    const int stride = gridDim.x * blockDim.x;
    for (; i < n4; i += stride) {
        out[i] = __builtin_convertvector(in[i], v4bf);
    }
}

// ---------------------------------------------------------------------------
// GEMM: C[M,N] = A[M,K] * B[N,K]^T  (both row-major, K-contiguous).
// 128 threads = 4 waves; block tile 128x128; wave tile 64x64.
// K staged in steps of 64 via TDM into double-buffered, row-padded LDS.
// MODE 0: bf16 row-major | MODE 1: bf16 per-head transpose | MODE 2: f32+bias
// ---------------------------------------------------------------------------
constexpr int LROW = 72;  // padded LDS row stride in elements (144 bytes)

template <int MODE>
__global__ __launch_bounds__(128) void gemm_nt(
    const __bf16* __restrict__ A, const __bf16* __restrict__ Bm,
    __bf16* __restrict__ Cb, float* __restrict__ Cf,
    const float* __restrict__ bias, int M, int N, int K) {
    const int lane   = threadIdx.x & 31;
    const int wv     = threadIdx.x >> 5;   // 0..3
    const int lanehi = lane >> 4;
    const int ln     = lane & 15;
    const int mb = blockIdx.y * 128;
    const int nb = blockIdx.x * 128;
    const int mw = (wv >> 1) * 64;         // wave row offset in block tile
    const int nw = (wv & 1) * 64;          // wave col offset in block tile

    __shared__ __bf16 sA[2][128 * LROW];
    __shared__ __bf16 sB[2][128 * LROW];

    v8f acc[4][4];
#pragma unroll
    for (int i = 0; i < 4; ++i)
#pragma unroll
        for (int j = 0; j < 4; ++j) acc[i][j] = {};

    const int steps = K / 64;

    auto stage = [&](int buf, int k) {
#if TDM_AVAILABLE
        if (wv == 0)
            tdm_load_2d((unsigned int)(uintptr_t)&sA[buf][0],
                        A + (size_t)mb * K + k, 64u, 128u,
                        (unsigned int)K, (unsigned int)M, (unsigned int)K, 4u);
        else if (wv == 1)
            tdm_load_2d((unsigned int)(uintptr_t)&sB[buf][0],
                        Bm + (size_t)nb * K + k, 64u, 128u,
                        (unsigned int)K, (unsigned int)N, (unsigned int)K, 4u);
#else
        // Fallback: cooperative synchronous staging (row per thread).
        {
            const int row = threadIdx.x;  // 0..127
            const __bf16* ga = A + (size_t)(mb + row) * K + k;
            const __bf16* gb = Bm + (size_t)(nb + row) * K + k;
#pragma unroll
            for (int c = 0; c < 8; ++c) {
                *(v8bf*)&sA[buf][row * LROW + c * 8] = *(const v8bf*)(ga + c * 8);
                *(v8bf*)&sB[buf][row * LROW + c * 8] = *(const v8bf*)(gb + c * 8);
            }
        }
#endif
    };

    stage(0, 0);  // prologue

    for (int i = 0; i < steps; ++i) {
        const int cur = i & 1;
        if (i + 1 < steps) {
            stage(cur ^ 1, (i + 1) * 64);
#if TDM_AVAILABLE
            if (wv < 2) __builtin_amdgcn_s_wait_tensorcnt(1);
#endif
        } else {
#if TDM_AVAILABLE
            if (wv < 2) __builtin_amdgcn_s_wait_tensorcnt(0);
#endif
        }
        __syncthreads();  // tiles for step i visible to all waves

#pragma unroll
        for (int kc = 0; kc < 2; ++kc) {
            v16bf af[4];
#pragma unroll
            for (int mt = 0; mt < 4; ++mt)
                af[mt] = load_a32(&sA[cur][(mw + mt * 16 + ln) * LROW + kc * 32],
                                  lanehi);
#pragma unroll
            for (int nt = 0; nt < 4; ++nt) {
                v16bf bf = load_b32(&sB[cur][(nw + nt * 16 + ln) * LROW + kc * 32],
                                    lanehi);
#pragma unroll
                for (int mt = 0; mt < 4; ++mt)
                    acc[mt][nt] = wmma_bf16(af[mt], bf, acc[mt][nt]);
            }
        }
        __syncthreads();  // step i consumed before buffer reuse
    }

#pragma unroll
    for (int mt = 0; mt < 4; ++mt)
#pragma unroll
        for (int nt = 0; nt < 4; ++nt)
#pragma unroll
            for (int r = 0; r < 8; ++r) {
                const int m = mb + mw + mt * 16 + r + lanehi * 8;
                const int n = nb + nw + nt * 16 + ln;
                const float v = acc[mt][nt][r];
                if (MODE == 0) {
                    Cb[(size_t)m * N + n] = (__bf16)v;
                } else if (MODE == 1) {
                    // m = b*2048 + s ; n = h*128 + hd -> vT[(b*16+h)*128+hd][s]
                    const int bb = m >> 11, s = m & 2047;
                    const int hh = n >> 7,  hd = n & 127;
                    Cb[((size_t)((bb * 16 + hh) * 128 + hd)) * 2048 + s] = (__bf16)v;
                } else {
                    Cf[(size_t)m * N + n] = v + bias[n];
                }
            }
}

// ---------------------------------------------------------------------------
// Flash attention (causal), per (b, h, 64-row q tile). 4 waves x 16 q rows.
// K tile (64x128) and V^T tile (128x64) are TDM-staged into double-buffered
// LDS and shared by all 4 waves (was 4x-redundant global streaming).
// scale = 1/sqrt(S) per reference.  O accumulated fp32, P via LDS transpose.
// ---------------------------------------------------------------------------
constexpr int KROW = 136;  // K tile LDS row stride in elements (272 bytes)
constexpr int VROW = 72;   // V tile LDS row stride in elements (144 bytes)

__global__ __launch_bounds__(128) void flash_attn(
    const __bf16* __restrict__ Q,   // [B*S][D]
    const __bf16* __restrict__ Km,  // [B*S][D]
    const __bf16* __restrict__ Vt,  // [B*H][HD][S]
    __bf16* __restrict__ Ctx)       // [B*S][D]
{
    constexpr int S = 2048, D = 2048, HD = 128;
    const float isc = 0.022097086912079612f;  // 1/sqrt(2048)

    const int lane   = threadIdx.x & 31;
    const int wv     = threadIdx.x >> 5;  // 0..3
    const int lanehi = lane >> 4;
    const int ln     = lane & 15;
    const int bx = blockIdx.x;            // 64-row q tile index
    const int h  = blockIdx.y;
    const int b  = blockIdx.z;
    const int qrow0 = bx * 64 + wv * 16;

    __shared__ __bf16 sK[2][64 * KROW];   // 34,816 B
    __shared__ __bf16 sV[2][128 * VROW];  // 36,864 B
    __shared__ __bf16 lp[4][16][64];      // per-wave P staging (8 KB)

    const __bf16* Kbase = Km + (size_t)(b * S) * D + h * HD;
    const __bf16* Vbase = Vt + (size_t)(b * 16 + h) * HD * S;

    auto stageKV = [&](int buf, int kt2) {
#if TDM_AVAILABLE
        if (wv == 0)  // K tile: 64 rows x 128 hd elems, global row stride D
            tdm_load_2d((unsigned int)(uintptr_t)&sK[buf][0],
                        Kbase + (size_t)(kt2 * 64) * D, 128u, 64u,
                        (unsigned int)D, (unsigned int)S, (unsigned int)D, 5u);
        else if (wv == 1)  // V^T tile: 128 hd rows x 64 s elems, row stride S
            tdm_load_2d((unsigned int)(uintptr_t)&sV[buf][0],
                        Vbase + kt2 * 64, 64u, 128u,
                        (unsigned int)S, (unsigned int)HD, (unsigned int)S, 4u);
#else
        {   // Fallback: cooperative synchronous staging.
            const int t = threadIdx.x;           // 0..127
            const int kr = t >> 1, kh = (t & 1) * 64;  // K: row, col-half
            const __bf16* gk = Kbase + (size_t)(kt2 * 64 + kr) * D + kh;
#pragma unroll
            for (int c = 0; c < 8; ++c)
                *(v8bf*)&sK[buf][kr * KROW + kh + c * 8] = *(const v8bf*)(gk + c * 8);
            const __bf16* gv = Vbase + (size_t)t * S + kt2 * 64;
#pragma unroll
            for (int c = 0; c < 8; ++c)
                *(v8bf*)&sV[buf][t * VROW + c * 8] = *(const v8bf*)(gv + c * 8);
        }
#endif
    };

    // Resident Q fragments: 16 rows x 128 (4 k-chunks of 32)
    const __bf16* qrow = Q + (size_t)(b * S + qrow0 + ln) * D + h * HD;
    v16bf qf[4];
#pragma unroll
    for (int kc = 0; kc < 4; ++kc) qf[kc] = load_a32(qrow + kc * 32, lanehi);

    v8f o[8];
#pragma unroll
    for (int i = 0; i < 8; ++i) o[i] = {};
    float mrow[8], lrow[8];
#pragma unroll
    for (int r = 0; r < 8; ++r) { mrow[r] = -1e30f; lrow[r] = 0.0f; }

    stageKV(0, 0);  // prologue

    for (int kt = 0; kt <= bx; ++kt) {
        const int cur = kt & 1;
        if (kt < bx) {
            stageKV(cur ^ 1, kt + 1);
#if TDM_AVAILABLE
            if (wv < 2) __builtin_amdgcn_s_wait_tensorcnt(1);
#endif
        } else {
#if TDM_AVAILABLE
            if (wv < 2) __builtin_amdgcn_s_wait_tensorcnt(0);
#endif
        }
        __syncthreads();  // K/V tiles for step kt visible to all waves

        // ---- scores S = Q * K^T  (16 x 64 per wave, K tile from LDS) ----
        v8f sacc[4];
#pragma unroll
        for (int nt = 0; nt < 4; ++nt) sacc[nt] = {};
#pragma unroll
        for (int nt = 0; nt < 4; ++nt) {
#pragma unroll
            for (int kc = 0; kc < 4; ++kc) {
                v16bf kf = load_b32(&sK[cur][(nt * 16 + ln) * KROW + kc * 32],
                                    lanehi);
                sacc[nt] = wmma_bf16(qf[kc], kf, sacc[nt]);
            }
        }

        // ---- causal mask on diagonal tile ----
        if (kt == bx) {
#pragma unroll
            for (int nt = 0; nt < 4; ++nt)
#pragma unroll
                for (int r = 0; r < 8; ++r) {
                    const int c = nt * 16 + ln;                 // col in 64-block
                    const int rr = wv * 16 + r + lanehi * 8;    // row in 64-block
                    if (c > rr) sacc[nt][r] = -1e30f;
                }
        }

        // ---- online softmax: row max (shfl within 16-lane half) ----
        float corr[8];
#pragma unroll
        for (int r = 0; r < 8; ++r) {
            float x = fmaxf(fmaxf(sacc[0][r], sacc[1][r]),
                            fmaxf(sacc[2][r], sacc[3][r]));
            x = fmaxf(x, __shfl_xor(x, 1));
            x = fmaxf(x, __shfl_xor(x, 2));
            x = fmaxf(x, __shfl_xor(x, 4));
            x = fmaxf(x, __shfl_xor(x, 8));
            const float mn = fmaxf(mrow[r], x);
            corr[r] = __expf((mrow[r] - mn) * isc);
            mrow[r] = mn;
            lrow[r] *= corr[r];
        }
#pragma unroll
        for (int n2 = 0; n2 < 8; ++n2)
#pragma unroll
            for (int r = 0; r < 8; ++r) o[n2][r] *= corr[r];

        // ---- P = exp((s-m)*isc); stage to LDS row-major bf16 ----
#pragma unroll
        for (int nt = 0; nt < 4; ++nt)
#pragma unroll
            for (int r = 0; r < 8; ++r) {
                const float p = __expf((sacc[nt][r] - mrow[r]) * isc);
                sacc[nt][r] = p;
                lp[wv][r + lanehi * 8][nt * 16 + ln] = (__bf16)p;
            }
#pragma unroll
        for (int r = 0; r < 8; ++r) {
            float s = (sacc[0][r] + sacc[1][r]) + (sacc[2][r] + sacc[3][r]);
            s += __shfl_xor(s, 1);
            s += __shfl_xor(s, 2);
            s += __shfl_xor(s, 4);
            s += __shfl_xor(s, 8);
            lrow[r] += s;
        }
        __syncthreads();  // uniform trip count -> converged

        // ---- O += P * V  (P in A-layout, V^T tile from LDS) ----
        v16bf pf[2];
#pragma unroll
        for (int kc2 = 0; kc2 < 2; ++kc2)
            pf[kc2] = load_a32(&lp[wv][ln][0] + kc2 * 32, lanehi);
#pragma unroll
        for (int n2 = 0; n2 < 8; ++n2) {
#pragma unroll
            for (int kc2 = 0; kc2 < 2; ++kc2) {
                v16bf vf = load_b32(&sV[cur][(n2 * 16 + ln) * VROW + kc2 * 32],
                                    lanehi);
                o[n2] = wmma_bf16(pf[kc2], vf, o[n2]);
            }
        }
        __syncthreads();  // step kt consumed before its buffer is re-filled
    }

    // ---- normalize and store ctx (bf16) ----
#pragma unroll
    for (int r = 0; r < 8; ++r) {
        const float inv = 1.0f / lrow[r];
        const int m = qrow0 + r + lanehi * 8;
#pragma unroll
        for (int n2 = 0; n2 < 8; ++n2) {
            Ctx[(size_t)(b * S + m) * D + h * HD + n2 * 16 + ln] =
                (__bf16)(o[n2][r] * inv);
        }
    }
}

// ---------------------------------------------------------------------------
// Host launcher
// ---------------------------------------------------------------------------
extern "C" void kernel_launch(void* const* d_in, const int* in_sizes, int n_in,
                              void* d_out, int out_size, void* d_ws, size_t ws_size,
                              hipStream_t stream) {
    (void)in_sizes; (void)n_in; (void)out_size; (void)ws_size;
    constexpr int B = 2, S = 2048, D = 2048;
    constexpr int M = B * S;                  // 4096
    constexpr size_t XN = (size_t)M * D;      // 8,388,608
    constexpr size_t WN = (size_t)D * D;      // 4,194,304

    const float* x_f  = (const float*)d_in[0];
    const float* wq_f = (const float*)d_in[1];
    const float* wk_f = (const float*)d_in[2];
    const float* wv_f = (const float*)d_in[3];
    const float* wo_f = (const float*)d_in[4];
    const float* bo_f = (const float*)d_in[5];
    float* out_f = (float*)d_out;

    __bf16* p = (__bf16*)d_ws;
    __bf16* x_bf  = p; p += XN;
    __bf16* wq_bf = p; p += WN;
    __bf16* wk_bf = p; p += WN;
    __bf16* wv_bf = p; p += WN;
    __bf16* wo_bf = p; p += WN;
    __bf16* q_bf  = p; p += XN;
    __bf16* k_bf  = p; p += XN;
    __bf16* vT_bf = p; p += XN;               // [B*H][128][S]
    __bf16* c_bf  = p; p += XN;

    // 1) fp32 -> bf16 conversions
    const int cvB = 2048, cvT = 256;
    cvt_f32_bf16<<<cvB, cvT, 0, stream>>>((const v4f*)x_f,  (v4bf*)x_bf,  (int)(XN / 4));
    cvt_f32_bf16<<<cvB, cvT, 0, stream>>>((const v4f*)wq_f, (v4bf*)wq_bf, (int)(WN / 4));
    cvt_f32_bf16<<<cvB, cvT, 0, stream>>>((const v4f*)wk_f, (v4bf*)wk_bf, (int)(WN / 4));
    cvt_f32_bf16<<<cvB, cvT, 0, stream>>>((const v4f*)wv_f, (v4bf*)wv_bf, (int)(WN / 4));
    cvt_f32_bf16<<<cvB, cvT, 0, stream>>>((const v4f*)wo_f, (v4bf*)wo_bf, (int)(WN / 4));

    // 2) Q/K/V projections (y = x * W^T)
    dim3 ggrid(D / 128, M / 128);  // (16, 32)
    dim3 gblk(128);
    gemm_nt<0><<<ggrid, gblk, 0, stream>>>(x_bf, wq_bf, q_bf,  nullptr, nullptr, M, D, D);
    gemm_nt<0><<<ggrid, gblk, 0, stream>>>(x_bf, wk_bf, k_bf,  nullptr, nullptr, M, D, D);
    gemm_nt<1><<<ggrid, gblk, 0, stream>>>(x_bf, wv_bf, vT_bf, nullptr, nullptr, M, D, D);

    // 3) causal flash attention
    dim3 fgrid(S / 64, 16, B);  // (32, 16, 2)
    flash_attn<<<fgrid, dim3(128), 0, stream>>>(q_bf, k_bf, vT_bf, c_bf);

    // 4) output projection + bias (fp32 out)
    gemm_nt<2><<<ggrid, gblk, 0, stream>>>(c_bf, wo_bf, nullptr, out_f, bo_f, M, D, D);
}
